// EffEnsembleDynamicModel_88888643158685
// MI455X (gfx1250) — compile-verified, matrix operand order checked by reading, not run on hardware
//
#include <hip/hip_runtime.h>
#include <hip/hip_bf16.h>

// ---------------- types ----------------
typedef __attribute__((ext_vector_type(16))) _Float16 v16h;
typedef __attribute__((ext_vector_type(8)))  float    v8f;
typedef __attribute__((ext_vector_type(4)))  unsigned int v4u;

union H16 { v16h h; v4u u[2]; };

// Load one WMMA f16 fragment (16 halves) as two 16-byte chunks.
// gfx1250 16-bit A/B layout: lane<16 holds K={0..7,16..23}, lane>=16 holds
// K={8..15,24..31} of its row/col -> element h<8 at base, h>=8 at base+16.
__device__ inline v16h frag16(const _Float16* p) {
  H16 r;
  r.u[0] = *(const v4u*)(p);
  r.u[1] = *(const v4u*)(p + 16);
  return r.h;
}

// CDNA5 async DMA: copy 16 bytes global -> LDS, tracked by ASYNCcnt.
// vdst VGPR holds the LDS byte offset, vaddr the 64-bit global address.
__device__ inline void async_copy16(unsigned lds_byte_off, const void* gaddr) {
  asm volatile("global_load_async_to_lds_b128 %0, %1, off"
               :: "v"(lds_byte_off), "v"(gaddr)
               : "memory");
}
__device__ inline void wait_async0() {
  asm volatile("s_wait_asynccnt 0" ::: "memory");
}

__device__ inline float softplusf(float x) {
  return (x > 20.f) ? x : __logf(1.f + __expf(x));
}

__device__ inline unsigned hashu(unsigned x) {
  x ^= x >> 16; x *= 0x7feb352dU;
  x ^= x >> 15; x *= 0x846ca68bU;
  x ^= x >> 16; return x;
}

#define NROWS   100000
#define RPM     20000      // rows per model (N / E_NUM)
#define EN      5           // elites
#define HP      256         // hidden 200 padded to 16*16
#define CP      64          // input 40 padded
#define OP      80          // 2*DOUT=66 padded to 5*16
#define DOUTV   33

// ---------------- weight prep: pad + transpose + f32->f16 ----------------
// w1t: [5][256][64]  (n-major, k contiguous)
// w2t,w3t: [5][256][256]
// w4t: [5][80][256]
__global__ void prep_weights(const float* __restrict__ W1, const float* __restrict__ W2,
                             const float* __restrict__ W3, const float* __restrict__ W4,
                             _Float16* __restrict__ w1t, _Float16* __restrict__ w2t,
                             _Float16* __restrict__ w3t, _Float16* __restrict__ w4t) {
  const int S1 = HP * CP;          // 16384
  const int S2 = HP * HP;          // 65536
  const int S4 = OP * HP;          // 20480
  const int T  = S1 + 2 * S2 + S4; // 167936 per model
  int idx = blockIdx.x * blockDim.x + threadIdx.x;
  if (idx >= EN * T) return;
  int mm = idx / T, rem = idx % T;
  if (rem < S1) {
    int nn = rem / CP, kk = rem % CP;
    float v = (nn < 200 && kk < 40) ? W1[mm * 40 * 200 + kk * 200 + nn] : 0.f;
    w1t[mm * S1 + rem] = (_Float16)v;
  } else if (rem < S1 + S2) {
    int rr = rem - S1; int nn = rr / HP, kk = rr % HP;
    float v = (nn < 200 && kk < 200) ? W2[mm * 200 * 200 + kk * 200 + nn] : 0.f;
    w2t[mm * S2 + rr] = (_Float16)v;
  } else if (rem < S1 + 2 * S2) {
    int rr = rem - S1 - S2; int nn = rr / HP, kk = rr % HP;
    float v = (nn < 200 && kk < 200) ? W3[mm * 200 * 200 + kk * 200 + nn] : 0.f;
    w3t[mm * S2 + rr] = (_Float16)v;
  } else {
    int rr = rem - S1 - 2 * S2; int nn = rr / HP, kk = rr % HP;
    float v = (nn < 66 && kk < 200) ? W4[mm * 200 * 66 + kk * 66 + nn] : 0.f;
    w4t[mm * S4 + rr] = (_Float16)v;
  }
}

// ---------------- fused layer: LDS acts x (global f16 weights) ----------------
// Each wave computes nct col-tiles x 4 row-tiles (64 rows per block).
__device__ inline void gemm_layer(const _Float16* __restrict__ in, int inStride, int kchunks,
                                  const _Float16* __restrict__ wt, int KP,
                                  const float* __restrict__ bias, int biasN,
                                  _Float16* __restrict__ outH, float* __restrict__ outF,
                                  bool doSwish, int ct0, int nct, int lane) {
  const int g  = lane >> 4;
  const int li = lane & 15;
  const v8f zero = {};
  v8f acc[2][4];
#pragma unroll
  for (int t = 0; t < 2; ++t)
#pragma unroll
    for (int r = 0; r < 4; ++r) acc[t][r] = zero;

  if (nct > 0) __builtin_prefetch(wt + (ct0 * 16 + li) * KP, 0, 1);

  for (int kc = 0; kc < kchunks; ++kc) {
    const int k0 = kc * 32;
    v16h a[4];
#pragma unroll
    for (int rt = 0; rt < 4; ++rt)
      a[rt] = frag16(in + (rt * 16 + li) * inStride + k0 + g * 8);
    for (int t = 0; t < nct; ++t) {
      const int n0 = (ct0 + t) * 16;
      v16h b = frag16(wt + (n0 + li) * KP + k0 + g * 8);
#pragma unroll
      for (int rt = 0; rt < 4; ++rt)
        acc[t][rt] = __builtin_amdgcn_wmma_f32_16x16x32_f16(
            false, a[rt], false, b, (short)0, acc[t][rt], false, false);
    }
  }

  // epilogue: D layout -> row = rt*16 + g*8 + v, col = ct*16 + li
  for (int t = 0; t < nct; ++t) {
    const int col = (ct0 + t) * 16 + li;
    const float bv = (col < biasN) ? bias[col] : 0.f;
#pragma unroll
    for (int rt = 0; rt < 4; ++rt) {
#pragma unroll
      for (int v = 0; v < 8; ++v) {
        float x = acc[t][rt][v] + bv;
        if (doSwish) x = x / (1.f + __expf(-x));
        const int row = rt * 16 + g * 8 + v;
        if (outF) outF[row * OP + col] = x;
        else      outH[row * HP + col] = (_Float16)x;
      }
    }
  }
}

// ---------------- fused ensemble MLP + sampling kernel ----------------
__global__ __launch_bounds__(256) void ensemble_mlp_kernel(
    const float* __restrict__ obs, const float* __restrict__ actn,
    const float* __restrict__ mu,  const float* __restrict__ sd,
    const _Float16* __restrict__ w1t, const _Float16* __restrict__ w2t,
    const _Float16* __restrict__ w3t, const _Float16* __restrict__ w4t,
    const float* __restrict__ b1, const float* __restrict__ b2,
    const float* __restrict__ b3, const float* __restrict__ b4,
    float* __restrict__ dout) {
  __shared__ _Float16 sA[64 * HP];  // 32 KB
  __shared__ _Float16 sB[64 * HP];  // 32 KB
  const int tid  = threadIdx.x;
  const int lane = tid & 31;
  const int wave = tid >> 5;
  const int m    = blockIdx.y;            // elite model (elites == arange(5))
  const int row0 = blockIdx.x * 64;

  // ---- stage raw obs/act rows into LDS via async DMA (ASYNCcnt path) ----
  // zstage: [64][40] f32 in sB region (10 KB). obs = 8x16B/row, act = 2x16B/row.
  float* zstage = (float*)sB;
  for (int t = tid; t < 64 * 8; t += 256) {  // observations
    int row = t >> 3, seg = t & 7;
    int gr = row0 + row;
    if (gr < RPM) {
      long n = (long)m * RPM + gr;
      async_copy16((unsigned)(size_t)(zstage + row * 40 + seg * 4),
                   obs + n * 32 + seg * 4);
    }
  }
  for (int t = tid; t < 64 * 2; t += 256) {  // actions
    int row = t >> 1, seg = t & 1;
    int gr = row0 + row;
    if (gr < RPM) {
      long n = (long)m * RPM + gr;
      async_copy16((unsigned)(size_t)(zstage + row * 40 + 32 + seg * 4),
                   actn + n * 8 + seg * 4);
    }
  }
  wait_async0();
  __syncthreads();

  // ---- normalize into f16 activation buffer sA [64][64], zero padded ----
  for (int idx = tid; idx < 64 * CP; idx += 256) {
    int row = idx >> 6, k = idx & 63;
    float v = 0.f;
    if (k < 40 && (row0 + row) < RPM)
      v = (zstage[row * 40 + k] - mu[k]) / sd[k];
    sA[idx] = (_Float16)v;
  }
  __syncthreads();

  // ---- 4 fused layers, activations ping-pong in LDS ----
  gemm_layer(sA, CP, 2, w1t + m * (HP * CP), CP, b1 + m * 200, 200, sB, nullptr, true, wave * 2, 2, lane);
  __syncthreads();
  gemm_layer(sB, HP, 7, w2t + m * (HP * HP), HP, b2 + m * 200, 200, sA, nullptr, true, wave * 2, 2, lane);
  __syncthreads();
  gemm_layer(sA, HP, 7, w3t + m * (HP * HP), HP, b3 + m * 200, 200, sB, nullptr, true, wave * 2, 2, lane);
  __syncthreads();
  float* outf = (float*)sA;  // layer-4 output: [64][80] f32 (20 KB, fits in sA)
  if (wave < EN)
    gemm_layer(sB, HP, 7, w4t + m * (OP * HP), HP, b4 + m * 66, 66, nullptr, outf, false, wave, 1, lane);
  __syncthreads();

  // ---- logvar clamp + gaussian sampling + outputs ----
  // nobuf: [64][32] f32 in sB region (8 KB) for the deterministic row-norm.
  float* nobuf = (float*)sB;
  float* out_nobs = dout;
  float* out_rew  = dout + (long)NROWS * 32;
  float* out_term = dout + (long)NROWS * 33;
  for (int idx = tid; idx < 64 * DOUTV; idx += 256) {
    int row = idx / DOUTV, j = idx % DOUTV;
    int gr = row0 + row;
    if (gr >= RPM) continue;
    long n = (long)m * RPM + gr;
    float mean = outf[row * OP + j];
    float lv   = outf[row * OP + DOUTV + j];
    lv = 0.5f  - softplusf(0.5f - lv);
    lv = -10.f + softplusf(lv + 10.f);
    float stdv = __expf(0.5f * lv);
    unsigned h1 = hashu((unsigned)(n * DOUTV + j));
    unsigned h2 = hashu(h1 ^ 0x9E3779B9u);
    float u1 = (float)(h1 | 1u) * 2.3283064365386963e-10f;
    float u2 = (float)h2 * 2.3283064365386963e-10f;
    float noise = sqrtf(-2.f * __logf(u1)) * __cosf(6.28318530718f * u2);
    float s = mean + noise * stdv;
    if (j < 32) {
      float nob = s + obs[n * 32 + j];
      out_nobs[n * 32 + j] = nob;
      nobuf[row * 32 + j] = nob;
    } else {
      out_rew[n] = s;  // REWARD_SCALE=1, BIAS=0
    }
  }
  __syncthreads();
  // deterministic terminal flag: sequential per-row sum of squares
  if (tid < 64) {
    int gr = row0 + tid;
    if (gr < RPM) {
      long n = (long)m * RPM + gr;
      float sum = 0.f;
#pragma unroll
      for (int j = 0; j < 32; ++j) {
        float v = nobuf[tid * 32 + j];
        sum += v * v;
      }
      out_term[n] = (sqrtf(sum) > 50.f) ? 1.f : 0.f;
    }
  }
}

extern "C" void kernel_launch(void* const* d_in, const int* in_sizes, int n_in,
                              void* d_out, int out_size, void* d_ws, size_t ws_size,
                              hipStream_t stream) {
  (void)in_sizes; (void)n_in; (void)out_size; (void)ws_size;
  const float* obs = (const float*)d_in[0];
  const float* act = (const float*)d_in[1];
  const float* mu  = (const float*)d_in[2];
  const float* sd  = (const float*)d_in[3];
  const float* W1  = (const float*)d_in[4];
  const float* b1  = (const float*)d_in[5];
  const float* W2  = (const float*)d_in[6];
  const float* b2  = (const float*)d_in[7];
  const float* W3  = (const float*)d_in[8];
  const float* b3  = (const float*)d_in[9];
  const float* W4  = (const float*)d_in[10];
  const float* b4  = (const float*)d_in[11];
  float* out = (float*)d_out;

  _Float16* w1t = (_Float16*)d_ws;                 // [5][256][64]
  _Float16* w2t = w1t + EN * (HP * CP);            // [5][256][256]
  _Float16* w3t = w2t + EN * (HP * HP);            // [5][256][256]
  _Float16* w4t = w3t + EN * (HP * HP);            // [5][80][256]

  const int totalW = EN * (HP * CP + 2 * HP * HP + OP * HP);
  prep_weights<<<(totalW + 255) / 256, 256, 0, stream>>>(W1, W2, W3, W4, w1t, w2t, w3t, w4t);

  dim3 grid((RPM + 63) / 64, EN, 1);
  ensemble_mlp_kernel<<<grid, 256, 0, stream>>>(obs, act, mu, sd,
                                                w1t, w2t, w3t, w4t,
                                                b1, b2, b3, b4, out);
}
